// MultiHeadSelfAttention2d_43310450213356
// MI455X (gfx1250) — matrix operand order repaired and checked by reading, not compiled, not run on hardware
//
#include <hip/hip_runtime.h>

#define Bn 2
#define Cn 256
#define NP 4096
#define OC3 768
#define HEADS 8
#define Dh 32

typedef unsigned int u32;
typedef unsigned short u16;
typedef u32 v4u __attribute__((ext_vector_type(4)));
typedef u32 v8u __attribute__((ext_vector_type(8)));
typedef float v8f __attribute__((ext_vector_type(8)));
typedef __bf16 v16bf __attribute__((ext_vector_type(16)));

// ---- helpers -------------------------------------------------------------

__device__ __forceinline__ u16 f2bf(float f) {
  u32 u = __builtin_bit_cast(u32, f);
  u32 lsb = (u >> 16) & 1u;
  u += 0x7fffu + lsb;  // round-to-nearest-even
  return (u16)(u >> 16);
}

__device__ __forceinline__ v8f wmma_bf16(v8u a, v8u b, v8f c) {
  return __builtin_amdgcn_wmma_f32_16x16x32_bf16(
      false, __builtin_bit_cast(v16bf, a),
      false, __builtin_bit_cast(v16bf, b),
      (short)0, c, false, false);
}

// A fragment: 16(M) x 32(K) bf16. Source row-major, K contiguous.
// Layout: lane l holds row M=l&15; VGPR i in [0,7]: K = 16g+2i+p,
// g = l>>4.  => one contiguous 32B run (two 16B chunks).
__device__ __forceinline__ v8u load_frag_a(const u16* base, int rowStride, int lane) {
  int m = lane & 15, g = (lane >> 4) & 1;
  const u16* p = base + (size_t)m * rowStride + g * 16;
  v4u lo = *(const v4u*)(p);
  v4u hi = *(const v4u*)(p + 8);
  return (v8u){lo[0], lo[1], lo[2], lo[3], hi[0], hi[1], hi[2], hi[3]};
}

// B fragment: 32(K) x 16(N) bf16. Source: for column n, the 32 K values are
// contiguous at base + n*colStride.  Lane l holds col N=l&15; K = 16g+2i+p.
__device__ __forceinline__ v8u load_frag_b(const u16* base, int colStride, int lane) {
  int nn = lane & 15, g = (lane >> 4) & 1;
  const u16* p = base + (size_t)nn * colStride + g * 16;
  v4u lo = *(const v4u*)(p);
  v4u hi = *(const v4u*)(p + 8);
  return (v8u){lo[0], lo[1], lo[2], lo[3], hi[0], hi[1], hi[2], hi[3]};
}

// ---- conversion kernels --------------------------------------------------

__global__ void cvt_bf16_kernel(const float* __restrict__ in, u16* __restrict__ out, int n) {
  int i = blockIdx.x * 256 + threadIdx.x;
  if (i < n) out[i] = f2bf(in[i]);
}

// x[b][c][n] f32 -> xT[b][n][c] bf16
__global__ void xpose_kernel(const float* __restrict__ x, u16* __restrict__ xT) {
  u32 i = blockIdx.x * 256 + threadIdx.x;  // ((b*Cn + c)*NP + n)
  int n = i & (NP - 1);
  u32 bc = i >> 12;
  int c = bc & (Cn - 1);
  int b = bc >> 8;
  xT[((size_t)b * NP + n) * Cn + c] = f2bf(x[i]);
}

// ---- QKV GEMM: out[n,o] = sum_c xT[b][n][c] * wqkv[o][c] -----------------
// 256 threads = 8 waves (4 M x 2 N); wave tile 32x32; block tile 128n x 64o.

__global__ __launch_bounds__(256) void qkv_gemm_kernel(
    const u16* __restrict__ xT, const u16* __restrict__ w,
    u16* __restrict__ qb, u16* __restrict__ kb, u16* __restrict__ vT) {
  int lane = threadIdx.x & 31;
  int wave = threadIdx.x >> 5;
  int wm = wave & 3, wn = wave >> 2;
  int bi = blockIdx.z;
  int n0 = blockIdx.x * 128 + wm * 32;
  int o0 = blockIdx.y * 64 + wn * 32;
  const u16* aBase = xT + (size_t)bi * NP * Cn;
  v8f c00 = {}, c01 = {}, c10 = {}, c11 = {};
  for (int kk = 0; kk < Cn; kk += 32) {
    v8u a0 = load_frag_a(aBase + (size_t)n0 * Cn + kk, Cn, lane);
    v8u a1 = load_frag_a(aBase + (size_t)(n0 + 16) * Cn + kk, Cn, lane);
    v8u b0 = load_frag_b(w + (size_t)o0 * Cn + kk, Cn, lane);
    v8u b1 = load_frag_b(w + (size_t)(o0 + 16) * Cn + kk, Cn, lane);
    c00 = wmma_bf16(a0, b0, c00);
    c01 = wmma_bf16(a0, b1, c01);
    c10 = wmma_bf16(a1, b0, c10);
    c11 = wmma_bf16(a1, b1, c11);
  }
  int g = (lane >> 4) & 1, ln = lane & 15;
#pragma unroll
  for (int ms = 0; ms < 2; ++ms)
#pragma unroll
    for (int ns = 0; ns < 2; ++ns) {
      v8f acc = ms == 0 ? (ns == 0 ? c00 : c01) : (ns == 0 ? c10 : c11);
#pragma unroll
      for (int r = 0; r < 8; ++r) {
        int n = n0 + ms * 16 + r + 8 * g;
        int o = o0 + ns * 16 + ln;
        u16 val = f2bf(acc[r]);
        int t = o >> 8, rem = o & 255;
        int head = rem >> 5, dd = rem & 31;
        size_t bh = (size_t)bi * HEADS + head;
        if (t == 0)      qb[(bh * NP + n) * Dh + dd] = val;
        else if (t == 1) kb[(bh * NP + n) * Dh + dd] = val;
        else             vT[(bh * Dh + dd) * NP + n] = val;  // V transposed
      }
    }
}

// ---- Flash attention: per (b,h), 64 query rows per block, 4 waves --------

__global__ __launch_bounds__(128) void attn_kernel(
    const u16* __restrict__ q, const u16* __restrict__ k,
    const u16* __restrict__ vT, u16* __restrict__ aT) {
  __shared__ __align__(16) u16 p_lds[4][16 * 64];  // per-wave 16x64 P tile
  int lane = threadIdx.x & 31;
  int wave = threadIdx.x >> 5;
  int bh = blockIdx.y;
  int b = bh >> 3, head = bh & 7;
  int qrow0 = blockIdx.x * 64 + wave * 16;
  int g = (lane >> 4) & 1, ln = lane & 15;
  const float scale = 0.1767766952966369f;  // 1/sqrt(32)

  // Q fragment: 16 rows x 32(K=d) -- single WMMA K-step
  v8u aq = load_frag_a(q + ((size_t)bh * NP + qrow0) * Dh, Dh, lane);

  float mrow[8], lrow[8];
#pragma unroll
  for (int r = 0; r < 8; ++r) { mrow[r] = -1e30f; lrow[r] = 0.f; }
  v8f o0 = {}, o1 = {};
  u16* pl = p_lds[wave];

  for (int kt = 0; kt < NP; kt += 64) {
    v8f z = {};
    v8f s0 = wmma_bf16(aq, load_frag_b(k + ((size_t)bh * NP + kt +  0) * Dh, Dh, lane), z);
    v8f s1 = wmma_bf16(aq, load_frag_b(k + ((size_t)bh * NP + kt + 16) * Dh, Dh, lane), z);
    v8f s2 = wmma_bf16(aq, load_frag_b(k + ((size_t)bh * NP + kt + 32) * Dh, Dh, lane), z);
    v8f s3 = wmma_bf16(aq, load_frag_b(k + ((size_t)bh * NP + kt + 48) * Dh, Dh, lane), z);
#pragma unroll
    for (int r = 0; r < 8; ++r) {
      float e0 = s0[r] * scale, e1 = s1[r] * scale;
      float e2 = s2[r] * scale, e3 = s3[r] * scale;
      float mx = fmaxf(fmaxf(e0, e1), fmaxf(e2, e3));
#pragma unroll
      for (int off = 1; off < 16; off <<= 1) mx = fmaxf(mx, __shfl_xor(mx, off, 32));
      float mnew = fmaxf(mrow[r], mx);
      float alpha = __expf(mrow[r] - mnew);
      float p0 = __expf(e0 - mnew), p1 = __expf(e1 - mnew);
      float p2 = __expf(e2 - mnew), p3 = __expf(e3 - mnew);
      float sum = p0 + p1 + p2 + p3;
#pragma unroll
      for (int off = 1; off < 16; off <<= 1) sum += __shfl_xor(sum, off, 32);
      lrow[r] = lrow[r] * alpha + sum;
      mrow[r] = mnew;
      o0[r] *= alpha;
      o1[r] *= alpha;
      int row = r + 8 * g;  // C-layout row owned by this lane group
      pl[row * 64 +  0 + ln] = f2bf(p0);
      pl[row * 64 + 16 + ln] = f2bf(p1);
      pl[row * 64 + 32 + ln] = f2bf(p2);
      pl[row * 64 + 48 + ln] = f2bf(p3);
    }
    asm volatile("s_wait_dscnt 0" ::: "memory");  // P stores -> A-frag loads
#pragma unroll
    for (int ks = 0; ks < 2; ++ks) {  // K(keys)=64 -> two 32-wide WMMA steps
      v8u ap  = load_frag_a(pl + ks * 32, 64, lane);
      v8u bv0 = load_frag_b(vT + ((size_t)bh * Dh +  0) * NP + kt + ks * 32, NP, lane);
      v8u bv1 = load_frag_b(vT + ((size_t)bh * Dh + 16) * NP + kt + ks * 32, NP, lane);
      o0 = wmma_bf16(ap, bv0, o0);
      o1 = wmma_bf16(ap, bv1, o1);
    }
  }
  // normalize + store as aT[b][n][head*32+dd] (ci contiguous for proj GEMM)
#pragma unroll
  for (int r = 0; r < 8; ++r) {
    int n = qrow0 + r + 8 * g;
    float inv = 1.f / lrow[r];
    size_t rowp = ((size_t)b * NP + n) * Cn + head * Dh;
    aT[rowp +  0 + ln] = f2bf(o0[r] * inv);
    aT[rowp + 16 + ln] = f2bf(o1[r] * inv);
  }
}

// ---- Proj GEMM + residual: out[b,co,n] = gamma*sum_ci aT[n,ci]*W[co,ci] + x

__global__ __launch_bounds__(256) void proj_gemm_kernel(
    const u16* __restrict__ aT, const u16* __restrict__ w,
    const float* __restrict__ x, const float* __restrict__ gamma,
    float* __restrict__ out) {
  int lane = threadIdx.x & 31;
  int wave = threadIdx.x >> 5;
  int wm = wave & 3, wn = wave >> 2;
  int bi = blockIdx.z;
  int n0 = blockIdx.x * 128 + wm * 32;
  int o0 = blockIdx.y * 64 + wn * 32;
  const u16* aBase = aT + (size_t)bi * NP * Cn;
  v8f c00 = {}, c01 = {}, c10 = {}, c11 = {};
  for (int kk = 0; kk < Cn; kk += 32) {
    v8u a0 = load_frag_a(aBase + (size_t)n0 * Cn + kk, Cn, lane);
    v8u a1 = load_frag_a(aBase + (size_t)(n0 + 16) * Cn + kk, Cn, lane);
    v8u b0 = load_frag_b(w + (size_t)o0 * Cn + kk, Cn, lane);
    v8u b1 = load_frag_b(w + (size_t)(o0 + 16) * Cn + kk, Cn, lane);
    c00 = wmma_bf16(a0, b0, c00);
    c01 = wmma_bf16(a0, b1, c01);
    c10 = wmma_bf16(a1, b0, c10);
    c11 = wmma_bf16(a1, b1, c11);
  }
  float gm = gamma[0];
  int g = (lane >> 4) & 1, ln = lane & 15;
#pragma unroll
  for (int ms = 0; ms < 2; ++ms)
#pragma unroll
    for (int ns = 0; ns < 2; ++ns) {
      v8f acc = ms == 0 ? (ns == 0 ? c00 : c01) : (ns == 0 ? c10 : c11);
#pragma unroll
      for (int r = 0; r < 8; ++r) {
        int n = n0 + ms * 16 + r + 8 * g;
        int co = o0 + ns * 16 + ln;
        size_t idx = ((size_t)bi * Cn + co) * NP + n;
        out[idx] = gm * acc[r] + x[idx];
      }
    }
}

// ---- host launcher -------------------------------------------------------

extern "C" void kernel_launch(void* const* d_in, const int* in_sizes, int n_in,
                              void* d_out, int out_size, void* d_ws, size_t ws_size,
                              hipStream_t stream) {
  (void)in_sizes; (void)n_in; (void)out_size; (void)ws_size;
  const float* x      = (const float*)d_in[0];  // [B, C, H, W] = [2,256,64,64]
  const float* qkv_w  = (const float*)d_in[1];  // [768, 256]
  const float* proj_w = (const float*)d_in[2];  // [256, 256]
  const float* gamma  = (const float*)d_in[3];  // [1]
  float* out = (float*)d_out;

  char* ws = (char*)d_ws;
  size_t off = 0;
  auto take = [&](size_t bytes) -> char* {
    char* p = ws + off;
    off = (off + bytes + 255) & ~(size_t)255;
    return p;
  };
  u16* xT    = (u16*)take((size_t)Bn * NP * Cn * 2);   // x transposed, bf16
  u16* wqkv  = (u16*)take((size_t)OC3 * Cn * 2);
  u16* wproj = (u16*)take((size_t)Cn * Cn * 2);
  u16* qb    = (u16*)take((size_t)Bn * HEADS * NP * Dh * 2);  // [b,h,n,d]
  u16* kb    = (u16*)take((size_t)Bn * HEADS * NP * Dh * 2);  // [b,h,n,d]
  u16* vT    = (u16*)take((size_t)Bn * HEADS * Dh * NP * 2);  // [b,h,d,n]
  u16* aT    = (u16*)take((size_t)Bn * NP * Cn * 2);          // [b,n,ci]

  cvt_bf16_kernel<<<(OC3 * Cn + 255) / 256, 256, 0, stream>>>(qkv_w, wqkv, OC3 * Cn);
  cvt_bf16_kernel<<<(Cn * Cn + 255) / 256, 256, 0, stream>>>(proj_w, wproj, Cn * Cn);
  xpose_kernel<<<(Bn * Cn * NP) / 256, 256, 0, stream>>>(x, xT);

  qkv_gemm_kernel<<<dim3(NP / 128, OC3 / 64, Bn), 256, 0, stream>>>(xT, wqkv, qb, kb, vT);
  attn_kernel<<<dim3(NP / 64, Bn * HEADS, 1), 128, 0, stream>>>(qb, kb, vT, aT);
  proj_gemm_kernel<<<dim3(NP / 128, Cn / 64, Bn), 256, 0, stream>>>(aT, wproj, x, gamma, out);
}
